// LinearAttention_39152921870872
// MI455X (gfx1250) — matrix operand order repaired and checked by reading, not compiled, not run on hardware
//
#include <hip/hip_runtime.h>
#include <math.h>

#define B_ 8
#define N_ 4096
#define M_ 4096
#define C_ 256
#define KSPLIT 8            // M split into 8 chunks of 512 for kv partials
#define MCHUNK (M_ / KSPLIT)

typedef __attribute__((ext_vector_type(16))) __bf16 v16bf;
typedef __attribute__((ext_vector_type(8)))  __bf16 v8bf;
typedef __attribute__((ext_vector_type(8)))  float  v8f;

union FragU { v16bf v; v8bf h[2]; };

// A fragment (16x32, MxK), row-major source, lane r=lane&15 holds row r.
// VGPRs 0-3: K = 8h..8h+7 ; VGPRs 4-7: K = 16+8h..16+8h+7  (h = lane>>4)
__device__ __forceinline__ v16bf load_a_frag(const __bf16* tile, int ld, int lane) {
    const int r = lane & 15;
    const int h = lane >> 4;
    const __bf16* p = tile + (size_t)r * ld + 8 * h;
    FragU u;
    u.h[0] = *(const v8bf*)(p);
    u.h[1] = *(const v8bf*)(p + 16);
    return u.v;
}

// B fragment (32x16, KxN) loaded from B-transposed storage (rows = N, contiguous K).
// Lane col n = lane&15; VGPRs 0-3: K = 8h..8h+7 ; VGPRs 4-7: K = 16+8h..16+8h+7  (h = lane>>4)
__device__ __forceinline__ v16bf load_b_fragT(const __bf16* tileT, int ld, int lane) {
    const int n  = lane & 15;
    const int kb = (lane >> 4) << 3;
    const __bf16* p = tileT + (size_t)n * ld + kb;
    FragU u;
    u.h[0] = *(const v8bf*)(p);
    u.h[1] = *(const v8bf*)(p + 16);
    return u.v;
}

// ---------------------------------------------------------------------------
// Pass 1: fp32 L2-normalize (optional) + convert to bf16, optional transpose.
// One block (256 threads) per row of length C=256.
// ---------------------------------------------------------------------------
__global__ __launch_bounds__(256)
void prep_kernel(const float* __restrict__ src, __bf16* __restrict__ dst,
                 int rowsPerBatch, int doNorm, int doTranspose) {
    const int tid = threadIdx.x;           // channel index 0..255
    const int row = blockIdx.x;            // 0 .. B*rowsPerBatch-1
    const float x = src[(size_t)row * C_ + tid];

    float scale = 1.0f;
    __shared__ float red[256];
    if (doNorm) {                           // uniform branch
        red[tid] = x * x;
        __syncthreads();
        #pragma unroll
        for (int off = 128; off > 0; off >>= 1) {
            if (tid < off) red[tid] += red[tid + off];
            __syncthreads();
        }
        scale = 1.0f / fmaxf(sqrtf(red[0]), 1e-12f);
    }
    const __bf16 y = (__bf16)(x * scale);

    if (doTranspose) {
        const int b = row / rowsPerBatch;
        const int m = row % rowsPerBatch;
        dst[(size_t)b * C_ * rowsPerBatch + (size_t)tid * rowsPerBatch + m] = y;
    } else {
        dst[(size_t)row * C_ + tid] = y;
    }
}

// ---------------------------------------------------------------------------
// Pass 2: kvT partials.  kvT[d][c] = sum_m vT[d][m] * kT[c][m]
// A = vT (row-major, ld=M), Bt = kT (row-major, ld=M).
// Block = 8 waves, block tile 128x128, wave tile 64x32 (4x2 WMMA tiles).
// grid = (C/128=2, C/128=2, B*KSPLIT)
// ---------------------------------------------------------------------------
__global__ __launch_bounds__(256)
void kv_kernel(const __bf16* __restrict__ vT, const __bf16* __restrict__ kT,
               float* __restrict__ part) {
    const int lane = threadIdx.x & 31;
    const int wave = threadIdx.x >> 5;                 // 0..7
    const int z    = blockIdx.z;
    const int b    = z / KSPLIT;
    const int s    = z % KSPLIT;
    const int d0   = blockIdx.x * 128 + (wave & 1) * 64;
    const int c0   = blockIdx.y * 128 + (wave >> 1) * 32;
    const int mBeg = s * MCHUNK;

    const __bf16* vTb = vT + (size_t)b * C_ * M_;
    const __bf16* kTb = kT + (size_t)b * C_ * M_;

    v8f zero = {};
    v8f acc[4][2];
    #pragma unroll
    for (int i = 0; i < 4; ++i)
        #pragma unroll
        for (int j = 0; j < 2; ++j) acc[i][j] = zero;

    for (int m0 = mBeg; m0 < mBeg + MCHUNK; m0 += 32) {
        v16bf a[4], bf[2];
        #pragma unroll
        for (int i = 0; i < 4; ++i)
            a[i] = load_a_frag(vTb + (size_t)(d0 + 16 * i) * M_ + m0, M_, lane);
        #pragma unroll
        for (int j = 0; j < 2; ++j)
            bf[j] = load_b_fragT(kTb + (size_t)(c0 + 16 * j) * M_ + m0, M_, lane);
        #pragma unroll
        for (int i = 0; i < 4; ++i)
            #pragma unroll
            for (int j = 0; j < 2; ++j)
                acc[i][j] = __builtin_amdgcn_wmma_f32_16x16x32_bf16(
                    false, a[i], false, bf[j], (short)0, acc[i][j], false, false);
    }

    // Store f32 partial: part[s][b][d][c]
    const int n    = lane & 15;
    const int mrow = (lane >> 4) * 8;
    float* outp = part + ((size_t)s * B_ + b) * C_ * C_;
    #pragma unroll
    for (int i = 0; i < 4; ++i)
        #pragma unroll
        for (int j = 0; j < 2; ++j)
            #pragma unroll
            for (int g = 0; g < 8; ++g)
                outp[(size_t)(d0 + 16 * i + mrow + g) * C_ + (c0 + 16 * j + n)] =
                    acc[i][j][g];
}

// ---------------------------------------------------------------------------
// Pass 3: reduce KSPLIT partials -> kvT bf16
// ---------------------------------------------------------------------------
__global__ __launch_bounds__(256)
void kv_reduce_kernel(const float* __restrict__ part, __bf16* __restrict__ kvT) {
    const size_t idx = (size_t)blockIdx.x * 256 + threadIdx.x;  // over B*C*C
    const size_t bcc = (size_t)B_ * C_ * C_;
    float s = 0.0f;
    #pragma unroll
    for (int k = 0; k < KSPLIT; ++k) s += part[(size_t)k * bcc + idx];
    kvT[idx] = (__bf16)s;
}

// ---------------------------------------------------------------------------
// Pass 4: ctx[n][d] = (1/N) * sum_c qn[n][c] * kvT[d][c]
// A = qn (row-major, ld=C), Bt = kvT (row-major, ld=C).
// grid = (N/128=32, C/128=2, B)
// ---------------------------------------------------------------------------
__global__ __launch_bounds__(256)
void ctx_kernel(const __bf16* __restrict__ qn, const __bf16* __restrict__ kvT,
                float* __restrict__ out) {
    const int lane = threadIdx.x & 31;
    const int wave = threadIdx.x >> 5;
    const int b    = blockIdx.z;
    const int n0   = blockIdx.x * 128 + (wave & 1) * 64;
    const int d0   = blockIdx.y * 128 + (wave >> 1) * 32;

    const __bf16* qb  = qn  + (size_t)b * N_ * C_;
    const __bf16* kvb = kvT + (size_t)b * C_ * C_;

    v8f zero = {};
    v8f acc[4][2];
    #pragma unroll
    for (int i = 0; i < 4; ++i)
        #pragma unroll
        for (int j = 0; j < 2; ++j) acc[i][j] = zero;

    #pragma unroll
    for (int c0 = 0; c0 < C_; c0 += 32) {
        v16bf a[4], bf[2];
        #pragma unroll
        for (int i = 0; i < 4; ++i)
            a[i] = load_a_frag(qb + (size_t)(n0 + 16 * i) * C_ + c0, C_, lane);
        #pragma unroll
        for (int j = 0; j < 2; ++j)
            bf[j] = load_b_fragT(kvb + (size_t)(d0 + 16 * j) * C_ + c0, C_, lane);
        #pragma unroll
        for (int i = 0; i < 4; ++i)
            #pragma unroll
            for (int j = 0; j < 2; ++j)
                acc[i][j] = __builtin_amdgcn_wmma_f32_16x16x32_bf16(
                    false, a[i], false, bf[j], (short)0, acc[i][j], false, false);
    }

    const float sc   = 1.0f / (float)N_;
    const int   n    = lane & 15;
    const int   mrow = (lane >> 4) * 8;
    float* ob = out + (size_t)b * N_ * C_;
    #pragma unroll
    for (int i = 0; i < 4; ++i)
        #pragma unroll
        for (int j = 0; j < 2; ++j)
            #pragma unroll
            for (int g = 0; g < 8; ++g)
                ob[(size_t)(n0 + 16 * i + mrow + g) * C_ + (d0 + 16 * j + n)] =
                    acc[i][j][g] * sc;
}

extern "C" void kernel_launch(void* const* d_in, const int* in_sizes, int n_in,
                              void* d_out, int out_size, void* d_ws, size_t ws_size,
                              hipStream_t stream) {
    (void)in_sizes; (void)n_in; (void)out_size; (void)ws_size;
    const float* q = (const float*)d_in[0];
    const float* k = (const float*)d_in[1];
    const float* v = (const float*)d_in[2];
    float* out = (float*)d_out;

    char* ws = (char*)d_ws;
    const size_t MB = 1024 * 1024;
    __bf16* qn   = (__bf16*)(ws + 0 * MB);    // 16 MB : (B,N,C) bf16
    __bf16* kT   = (__bf16*)(ws + 16 * MB);   // 16 MB : (B,C,M) bf16
    __bf16* vT   = (__bf16*)(ws + 32 * MB);   // 16 MB : (B,C,M) bf16
    float*  part = (float*) (ws + 48 * MB);   // 16 MB : (KSPLIT,B,C,C) f32
    __bf16* kvT  = (__bf16*)(ws + 64 * MB);   //  1 MB : (B,C,C) bf16

    // Pass 1: normalize/convert (q straight, k/v transposed)
    prep_kernel<<<B_ * N_, 256, 0, stream>>>(q, qn, N_, 1, 0);
    prep_kernel<<<B_ * M_, 256, 0, stream>>>(k, kT, M_, 1, 1);
    prep_kernel<<<B_ * M_, 256, 0, stream>>>(v, vT, M_, 0, 1);

    // Pass 2: kvT partials over M chunks
    dim3 g2(C_ / 128, C_ / 128, B_ * KSPLIT);
    kv_kernel<<<g2, 256, 0, stream>>>(vT, kT, part);

    // Pass 3: reduce partials -> bf16 kvT
    kv_reduce_kernel<<<(B_ * C_ * C_) / 256, 256, 0, stream>>>(part, kvT);

    // Pass 4: context = q * kv / N
    dim3 g3(N_ / 128, C_ / 128, B_);
    ctx_kernel<<<g3, 256, 0, stream>>>(qn, kvT, out);
}
